// VariationalLSTM_51427938402416
// MI455X (gfx1250) — matrix operand order repaired and checked by reading, not compiled, not run on hardware
//
#include <hip/hip_runtime.h>
#include <hip/hip_bf16.h>
#include <math.h>

typedef __attribute__((ext_vector_type(8)))  float  v8f;
typedef __attribute__((ext_vector_type(16))) __bf16 v16bf;

#define Bsz   128
#define Lseq  128
#define Idim  256
#define Hdim  256
#define Kdim  512          // I + H
#define G4H   1024         // 4*H
#define NBLK  64           // persistent grid: 4 row-groups x 16 col-tiles

__device__ __forceinline__ float sigmoidf_(float x) { return 1.0f / (1.0f + __expf(-x)); }

// ---------------------------------------------------------------- grid barrier
__device__ __forceinline__ void grid_barrier(unsigned* cnt, unsigned* sns, unsigned phase) {
    __threadfence();
    __syncthreads();
    if (threadIdx.x == 0) {
        unsigned prev = __hip_atomic_fetch_add(cnt, 1u, __ATOMIC_ACQ_REL, __HIP_MEMORY_SCOPE_AGENT);
        if (prev == NBLK - 1) {
            __hip_atomic_store(cnt, 0u, __ATOMIC_RELAXED, __HIP_MEMORY_SCOPE_AGENT);
            __hip_atomic_store(sns, phase, __ATOMIC_RELEASE, __HIP_MEMORY_SCOPE_AGENT);
        } else {
            while (__hip_atomic_load(sns, __ATOMIC_ACQUIRE, __HIP_MEMORY_SCOPE_AGENT) != phase)
                __builtin_amdgcn_s_sleep(1);
        }
    }
    __syncthreads();
    __threadfence();
}

// ------------------------------------------------------- one layer, one step
// Block covers batch rows [b0,b0+32), hidden cols [n0,n0+16), all 4 gates.
// Wave wv (0..3) computes gate wv's two 16x16 tiles via v_wmma_f32_16x16x32_bf16.
#define PACK8(vec, base, f0, f1)                                               \
    vec[(base)+0] = (__bf16)f0.x; vec[(base)+1] = (__bf16)f0.y;                \
    vec[(base)+2] = (__bf16)f0.z; vec[(base)+3] = (__bf16)f0.w;                \
    vec[(base)+4] = (__bf16)f1.x; vec[(base)+5] = (__bf16)f1.y;                \
    vec[(base)+6] = (__bf16)f1.z; vec[(base)+7] = (__bf16)f1.w;

__device__ __forceinline__ void layer_step(
    int b0, int n0,
    const float* __restrict__ zbase, size_t zstride,   // K < 256 source (row stride)
    const float* __restrict__ hread,                   // K >= 256 source (stride Hdim)
    const float* __restrict__ wmu,  const float* __restrict__ wstd,
    const float* __restrict__ epsw,                    // already offset by t
    const float* __restrict__ bmu,  const float* __restrict__ bstd,
    const float* __restrict__ epsb,                    // already offset by t
    float* __restrict__ cbuf, float* __restrict__ hwrite,
    float* __restrict__ outbase, size_t ostride,       // nullptr for layer 0
    float (*gl)[32][16])
{
    const int tid  = threadIdx.x;
    const int lane = tid & 31;
    const int wv   = tid >> 5;          // gate index: i,f,g,o
    const int hi   = lane >> 4;         // lane half (K-group select)
    const int l16  = lane & 15;         // A: M row; B: N col

    const int jRow = wv * Hdim + n0 + l16;   // row of w [4H x K] this lane samples
    v8f acc0 = {0.f,0.f,0.f,0.f,0.f,0.f,0.f,0.f};
    v8f acc1 = acc0;

    const int kBoff = hi << 4;   // B lane: 16 contiguous K at kc + 16*hi
    const int kAoff = hi << 4;   // A lane: 16 contiguous K at kc + 16*hi

    #pragma unroll 2
    for (int kc = 0; kc < Kdim; kc += 32) {
        // ---- B operand: sample w = mu + eps*std on the fly, convert to bf16
        size_t widx = (size_t)jRow * Kdim + (size_t)(kc + kBoff);
        const float4* mu4 = (const float4*)(wmu  + widx);
        const float4* sd4 = (const float4*)(wstd + widx);
        const float4* ep4 = (const float4*)(epsw + widx);
        v16bf bm;
        #pragma unroll
        for (int q = 0; q < 4; ++q) {
            float4 m = mu4[q], s = sd4[q], e = ep4[q];
            bm[4*q+0] = (__bf16)__builtin_fmaf(e.x, s.x, m.x);
            bm[4*q+1] = (__bf16)__builtin_fmaf(e.y, s.y, m.y);
            bm[4*q+2] = (__bf16)__builtin_fmaf(e.z, s.z, m.z);
            bm[4*q+3] = (__bf16)__builtin_fmaf(e.w, s.w, m.w);
        }

        // ---- A operands: z = [x_t | h_prev]; two 16-row tiles
        const float* src; size_t rstr; int kz;
        if (kc < Idim) { src = zbase; rstr = zstride; kz = kc; }
        else           { src = hread; rstr = Hdim;    kz = kc - Idim; }
        const int ka = kz + kAoff;
        const float4* p0 = (const float4*)(src + (size_t)(b0 +      l16) * rstr + ka);
        const float4* p1 = (const float4*)(src + (size_t)(b0 + 16 + l16) * rstr + ka);
        float4 u0 = p0[0], u1 = p0[1], u2 = p0[2], u3 = p0[3];   // k: +0..15
        float4 y0 = p1[0], y1 = p1[1], y2 = p1[2], y3 = p1[3];
        v16bf a0, a1;
        PACK8(a0, 0, u0, u1); PACK8(a0, 8, u2, u3);
        PACK8(a1, 0, y0, y1); PACK8(a1, 8, y2, y3);

        acc0 = __builtin_amdgcn_wmma_f32_16x16x32_bf16(false, a0, false, bm,
                                                       (short)0, acc0, false, false);
        acc1 = __builtin_amdgcn_wmma_f32_16x16x32_bf16(false, a1, false, bm,
                                                       (short)0, acc1, false, false);
    }

    // ---- sampled bias (per column, same for every row in the tile)
    {
        float bias = __builtin_fmaf(epsb[jRow], bstd[jRow], bmu[jRow]);
        #pragma unroll
        for (int r = 0; r < 8; ++r) { acc0[r] += bias; acc1[r] += bias; }
    }

    // ---- exchange gates through LDS: gl[gate][m 0..31][n 0..15]
    #pragma unroll
    for (int r = 0; r < 8; ++r) {
        gl[wv][     r + (hi << 3)][l16] = acc0[r];
        gl[wv][16 + r + (hi << 3)][l16] = acc1[r];
    }
    __syncthreads();

    // ---- elementwise LSTM cell: 512 elems / 128 threads = 4 each
    const int m  = tid >> 2;            // 0..31
    const int nq = (tid & 3) << 2;      // 0,4,8,12
    const int b  = b0 + m;
    #pragma unroll
    for (int q = 0; q < 4; ++q) {
        int n  = nq + q;
        int nh = n0 + n;
        float ig = sigmoidf_(gl[0][m][n]);
        float fg = sigmoidf_(gl[1][m][n]);
        float gg = tanhf    (gl[2][m][n]);
        float og = sigmoidf_(gl[3][m][n]);
        size_t ci = (size_t)b * Hdim + nh;
        float cn = __builtin_fmaf(fg, cbuf[ci], ig * gg);
        cbuf[ci]   = cn;
        float hn   = og * tanhf(cn);
        hwrite[ci] = hn;
        if (outbase) outbase[(size_t)b * ostride + nh] = hn;
    }
    __syncthreads();
}

// ------------------------------------------------------------ persistent LSTM
__global__ __launch_bounds__(128, 1) void lstm_persistent(
    const float* __restrict__ x,   float* __restrict__ out,
    const float* __restrict__ wmu0, const float* __restrict__ bmu0,
    const float* __restrict__ epsw0, const float* __restrict__ epsb0,
    const float* __restrict__ wstd0, const float* __restrict__ bstd0,
    const float* __restrict__ wmu1, const float* __restrict__ bmu1,
    const float* __restrict__ epsw1, const float* __restrict__ epsb1,
    const float* __restrict__ wstd1, const float* __restrict__ bstd1,
    float* h0a, float* h0b, float* c0,
    float* h1a, float* h1b, float* c1,
    unsigned* bar)
{
    __shared__ float gl[4][32][16];
    const int b0 = (blockIdx.x >> 4) * 32;   // 4 row-groups
    const int n0 = (blockIdx.x & 15) * 16;   // 16 col-tiles
    unsigned* cnt = bar;
    unsigned* sns = bar + 1;

    for (int t = 0; t < Lseq; ++t) {
        const float* h0r = (t & 1) ? h0b : h0a;
        float*       h0w = (t & 1) ? h0a : h0b;
        const float* h1r = (t & 1) ? h1b : h1a;
        float*       h1w = (t & 1) ? h1a : h1b;

        // layer 0: z = [x_t | h0_prev]
        layer_step(b0, n0,
                   x + (size_t)t * Idim, (size_t)Lseq * Idim, h0r,
                   wmu0, wstd0, epsw0 + (size_t)t * G4H * Kdim,
                   bmu0, bstd0, epsb0 + (size_t)t * G4H,
                   c0, h0w, nullptr, 0, gl);
        grid_barrier(cnt, sns, 2u * (unsigned)t + 1u);

        // layer 1: z = [h0_new | h1_prev]; writes output[b, t, :]
        layer_step(b0, n0,
                   h0w, (size_t)Hdim, h1r,
                   wmu1, wstd1, epsw1 + (size_t)t * G4H * Kdim,
                   bmu1, bstd1, epsb1 + (size_t)t * G4H,
                   c1, h1w, out + (size_t)t * Hdim, (size_t)Lseq * Hdim, gl);
        grid_barrier(cnt, sns, 2u * (unsigned)t + 2u);
    }
}

// ------------------------------------------------------------------- prologue
__global__ void prep(const float* wlv0, const float* wlv1,
                     const float* blv0, const float* blv1,
                     float* wstd0, float* wstd1, float* bstd0, float* bstd1,
                     float* states, unsigned* bar)
{
    size_t stride = (size_t)gridDim.x * blockDim.x;
    size_t i0 = (size_t)blockIdx.x * blockDim.x + threadIdx.x;
    for (size_t i = i0; i < (size_t)G4H * Kdim; i += stride) {
        wstd0[i] = __expf(0.5f * wlv0[i]);
        wstd1[i] = __expf(0.5f * wlv1[i]);
    }
    for (size_t i = i0; i < (size_t)G4H; i += stride) {
        bstd0[i] = __expf(0.5f * blv0[i]);
        bstd1[i] = __expf(0.5f * blv1[i]);
    }
    for (size_t i = i0; i < (size_t)6 * Bsz * Hdim; i += stride) states[i] = 0.0f;
    if (i0 == 0) { bar[0] = 0u; bar[1] = 0u; }
}

// ----------------------------------------------------------- KL partial sums
__global__ void kl_partials(const float* wmu0, const float* wlv0,
                            const float* bmu0, const float* blv0,
                            const float* wmu1, const float* wlv1,
                            const float* bmu1, const float* blv1,
                            float* part)
{
    const size_t W = (size_t)G4H * Kdim;    // 524288
    const size_t N = 2 * W + 2 * G4H;       // 1050624
    float acc = 0.0f;
    size_t stride = (size_t)gridDim.x * blockDim.x;
    for (size_t i = (size_t)blockIdx.x * blockDim.x + threadIdx.x; i < N; i += stride) {
        const float *mu, *lv; size_t j = i;
        if (j < W)                { mu = wmu0; lv = wlv0; }
        else if ((j -= W) < G4H)  { mu = bmu0; lv = blv0; }
        else if ((j -= G4H) < W)  { mu = wmu1; lv = wlv1; }
        else { j -= W;              mu = bmu1; lv = blv1; }
        float m = mu[j], l = lv[j];
        acc += m * m + __expf(l) - l;       // prior_std = 1
    }
    __shared__ float red[256];
    red[threadIdx.x] = acc;
    __syncthreads();
    for (int s = 128; s > 0; s >>= 1) {
        if (threadIdx.x < s) red[threadIdx.x] += red[threadIdx.x + s];
        __syncthreads();
    }
    if (threadIdx.x == 0) part[blockIdx.x] = red[0];
}

// ------------------------------------------------------------------- epilogue
__global__ void epilogue(const float* h0, const float* h1,
                         const float* c0, const float* c1,
                         const float* klp, float* dout)
{
    const size_t OUT = (size_t)Bsz * Lseq * Hdim;      // 4194304
    const size_t S   = (size_t)Bsz * Hdim;             // 32768
    size_t idx = (size_t)blockIdx.x * blockDim.x + threadIdx.x;
    if (idx < S) {
        dout[OUT + idx]         = h0[idx];             // h_final[0]
        dout[OUT + S + idx]     = h1[idx];             // h_final[1]
        dout[OUT + 2 * S + idx] = c0[idx];             // c_final[0]
        dout[OUT + 3 * S + idx] = c1[idx];             // c_final[1]
    }
    if (idx == 0) {
        float s = 0.0f;
        for (int i = 0; i < 256; ++i) s += klp[i];
        dout[OUT + 4 * S] = 0.5f * s;                  // total_kl
    }
}

// --------------------------------------------------------------------- launch
extern "C" void kernel_launch(void* const* d_in, const int* in_sizes, int n_in,
                              void* d_out, int out_size, void* d_ws, size_t ws_size,
                              hipStream_t stream)
{
    const float* x     = (const float*)d_in[0];
    const float* wmu0  = (const float*)d_in[1];
    const float* wlv0  = (const float*)d_in[2];
    const float* bmu0  = (const float*)d_in[3];
    const float* blv0  = (const float*)d_in[4];
    const float* wmu1  = (const float*)d_in[5];
    const float* wlv1  = (const float*)d_in[6];
    const float* bmu1  = (const float*)d_in[7];
    const float* blv1  = (const float*)d_in[8];
    const float* epsw0 = (const float*)d_in[9];
    const float* epsb0 = (const float*)d_in[10];
    const float* epsw1 = (const float*)d_in[11];
    const float* epsb1 = (const float*)d_in[12];
    float* out = (float*)d_out;

    // workspace layout (floats); ~5 MB required
    float* ws    = (float*)d_ws;
    float* wstd0 = ws;                       // 524288
    float* wstd1 = ws + 524288;              // 524288
    float* bstd0 = ws + 1048576;             // 1024
    float* bstd1 = ws + 1049600;             // 1024
    float* st    = ws + 1050624;             // 6 * 32768 state buffers
    float* h0a = st;            float* h0b = st + 32768;  float* c0 = st + 65536;
    float* h1a = st + 98304;    float* h1b = st + 131072; float* c1 = st + 163840;
    float* klp = ws + 1247232;               // 256 partials
    unsigned* bar = (unsigned*)(ws + 1247488);

    prep<<<512, 256, 0, stream>>>(wlv0, wlv1, blv0, blv1,
                                  wstd0, wstd1, bstd0, bstd1, st, bar);
    kl_partials<<<256, 256, 0, stream>>>(wmu0, wlv0, bmu0, blv0,
                                         wmu1, wlv1, bmu1, blv1, klp);
    lstm_persistent<<<NBLK, 128, 0, stream>>>(x, out,
                                              wmu0, bmu0, epsw0, epsb0, wstd0, bstd0,
                                              wmu1, bmu1, epsw1, epsb1, wstd1, bstd1,
                                              h0a, h0b, c0, h1a, h1b, c1, bar);
    // after 128 steps (even), final h states live in the 'a' buffers
    epilogue<<<128, 256, 0, stream>>>(h0a, h1a, c0, c1, klp, out);
}
